// get_model_75754633167350
// MI455X (gfx1250) — matrix operand order repaired and checked by reading, not compiled, not run on hardware
//
#include <hip/hip_runtime.h>

typedef _Float16 f16;
typedef __attribute__((ext_vector_type(16))) _Float16 v16h;
typedef __attribute__((ext_vector_type(8)))  _Float16 v8h;
typedef __attribute__((ext_vector_type(8)))  float    v8f;

#define NB 8
#define NP 2048
#define DV 256
#define NPT 128
#define NSAMP 64

#define WMMA(a,b,c) __builtin_amdgcn_wmma_f32_16x16x32_f16(false,(a),false,(b),(short)0,(c),false,false)
#define CAT16(lo,hi) __builtin_shufflevector((lo),(hi),0,1,2,3,4,5,6,7,8,9,10,11,12,13,14,15)

// ---------------- elementwise / setup kernels ----------------

__global__ void k_transpose(const float* __restrict__ xyz, float* __restrict__ pts) {
  int i = blockIdx.x * blockDim.x + threadIdx.x;   // NB*NP
  if (i >= NB * NP) return;
  int b = i / NP, n = i % NP;
  for (int c = 0; c < 3; ++c)
    pts[(b * NP + n) * 3 + c] = xyz[(b * 3 + c) * NP + n];
}

__global__ void k_eA(const float* __restrict__ pts, const float* __restrict__ A,
                     float* __restrict__ eAr, float* __restrict__ eAi,
                     f16* __restrict__ eArh, f16* __restrict__ eAih) {
  int i = blockIdx.x * blockDim.x + threadIdx.x;   // NB*NP*DV
  if (i >= NB * NP * DV) return;
  int d = i & (DV - 1);
  int bn = i >> 8;
  const float* p = pts + bn * 3;
  float ph = p[0] * A[d] + p[1] * A[DV + d] + p[2] * A[2 * DV + d];
  float s, c;
  __sincosf(ph, &s, &c);
  eAr[i] = c; eAi[i] = s;
  eArh[i] = (f16)c; eAih[i] = (f16)s;
}

// weight transpose + f32->f16 convert with K zero-padding: W(N,K) -> WT(Kpad,N)
__global__ void k_wT(const float* __restrict__ W, f16* __restrict__ WT,
                     int N, int K, int Kpad) {
  int i = blockIdx.x * blockDim.x + threadIdx.x;
  if (i >= Kpad * N) return;
  int k = i / N, n = i % N;
  WT[i] = (k < K) ? (f16)W[n * K + k] : (f16)0.0f;
}

// ---------------- veckm: fused-J WMMA GEMM  M = J @ eA ----------------
// grid (NP/16 row tiles, 2 col halves, NB), block 256 (8 waves, each 16 cols)
__launch_bounds__(256)
__global__ void k_veckm_gemm(const float* __restrict__ pts,
                             const f16* __restrict__ eArh, const f16* __restrict__ eAih,
                             float* __restrict__ Mr, float* __restrict__ Mi) {
  __shared__ float sP[32][3];
  __shared__ __attribute__((aligned(16))) f16 sErt[128][32];  // transposed (n, k)
  __shared__ __attribute__((aligned(16))) f16 sEit[128][32];
  int b = blockIdx.z, ch = blockIdx.y, rt = blockIdx.x;
  int tid = threadIdx.x;
  int wave = tid >> 5, lane = tid & 31;
  int half = lane >> 4, lm = lane & 15;
  int row = rt * 16 + lm;
  const float* prow = pts + ((size_t)b * NP + row) * 3;
  float px = prow[0], py = prow[1], pz = prow[2];
  int ncl = wave * 16 + lm;          // column within the 128-wide slice
  int ncol = ch * 128 + ncl;         // global column
  v8f cr = {}; v8f ci = {};
  for (int k0 = 0; k0 < NP; k0 += 32) {
    __syncthreads();
    if (tid < 96) {
      int kk = tid / 3, c = tid % 3;
      sP[kk][c] = pts[((size_t)b * NP + k0 + kk) * 3 + c];
    }
    // stage eA tile (32 rows(k) x 128 cols(n)) transposed into LDS, 8-half vectors
    for (int i = tid; i < (32 * 128) / 8; i += 256) {
      int idx = i * 8;
      int kk = idx >> 7, c = idx & 127;
      size_t g = ((size_t)(b * NP + k0 + kk)) * DV + ch * 128 + c;
      v8h vr = *(const v8h*)&eArh[g];
      v8h vi = *(const v8h*)&eAih[g];
#pragma unroll
      for (int j = 0; j < 8; ++j) {
        sErt[c + j][kk] = vr[j];
        sEit[c + j][kk] = vi[j];
      }
    }
    __syncthreads();
    // A-fragment: J values computed on the fly (16-bit A 16x32 layout)
    v16h af = {};
#pragma unroll
    for (int e = 0; e < 16; ++e) {
      int ka = e + half * 8 + ((e >= 8) ? 8 : 0);
      float dx = px - sP[ka][0], dy = py - sP[ka][1], dz = pz - sP[ka][2];
      af[e] = (f16)__expf(-18.0f * (dx * dx + dy * dy + dz * dz));
    }
    // B-fragments: 8-half runs at k = 8h and 16 + 8h from transposed LDS (wide ds loads)
    v8h rlo = *(const v8h*)&sErt[ncl][half * 8];
    v8h rhi = *(const v8h*)&sErt[ncl][16 + half * 8];
    v8h ilo = *(const v8h*)&sEit[ncl][half * 8];
    v8h ihi = *(const v8h*)&sEit[ncl][16 + half * 8];
    v16h brf = CAT16(rlo, rhi);
    v16h bif = CAT16(ilo, ihi);
    cr = WMMA(af, brf, cr);
    ci = WMMA(af, bif, ci);
  }
#pragma unroll
  for (int r = 0; r < 8; ++r) {
    int m = rt * 16 + r + (half ? 8 : 0);            // f32 C/D layout
    size_t o = ((size_t)(b * NP + m)) * DV + ncol;
    Mr[o] = cr[r];
    Mi[o] = ci[r];
  }
}

// G = (M * conj(eA)); row-normalize to norm SQRT_D=16; emit f16
__global__ void k_gnorm(const float* __restrict__ Mr, const float* __restrict__ Mi,
                        const float* __restrict__ eAr, const float* __restrict__ eAi,
                        f16* __restrict__ Grh, f16* __restrict__ Gih) {
  __shared__ float red[256];
  int bn = blockIdx.x, d = threadIdx.x;
  size_t o = (size_t)bn * DV + d;
  float mr = Mr[o], mi = Mi[o], er = eAr[o], ei = eAi[o];
  float gr = mr * er + mi * ei;
  float gi = mi * er - mr * ei;
  red[d] = gr * gr + gi * gi;
  __syncthreads();
  for (int s = 128; s > 0; s >>= 1) {
    if (d < s) red[d] += red[d + s];
    __syncthreads();
  }
  float scale = 16.0f * rsqrtf(red[0] + 1e-30f);
  Grh[o] = (f16)(gr * scale);
  Gih[o] = (f16)(gi * scale);
}

// ---------------- generic LDS-staged f16 WMMA GEMM ----------------
// C(MxN) {=, +=, -=} A(MxK) @ B(KxN); M%32==0, N%64==0, K%32==0
// grid (N/64, M/32), block 256 (8 waves, 2x4 tile grid)
__launch_bounds__(256)
__global__ void k_gemm(const f16* __restrict__ A, const f16* __restrict__ B,
                       float* __restrict__ C, int M, int N, int K, int mode) {
  __shared__ __attribute__((aligned(16))) f16 sA[32][32];    // row-major (m, k)
  __shared__ __attribute__((aligned(16))) f16 sBt[64][32];   // transposed (n, k)
  int tid = threadIdx.x, wave = tid >> 5, lane = tid & 31;
  int wr = wave >> 2, wc = wave & 3;
  int half = lane >> 4, lm = lane & 15;
  int m0 = blockIdx.y * 32, n0 = blockIdx.x * 64;
  int arow = wr * 16 + lm;
  int bcol = wc * 16 + lm;
  // staging coordinates (fixed per thread)
  int aidx = tid * 4, ar = aidx >> 5, ac = aidx & 31;        // 1024 halves of A
  int bidx = tid * 8, brw = bidx >> 6, bcl = bidx & 63;      // 2048 halves of B
  v8f acc = {};
  for (int k0 = 0; k0 < K; k0 += 32) {
    __syncthreads();
    *(uint2*)&sA[ar][ac] = *(const uint2*)&A[(size_t)(m0 + ar) * K + k0 + ac];
    v8h bv = *(const v8h*)&B[(size_t)(k0 + brw) * N + n0 + bcl];
#pragma unroll
    for (int j = 0; j < 8; ++j) sBt[bcl + j][brw] = bv[j];
    if (k0 + 32 < K) {  // prefetch next K tile (global_prefetch_b8, speculative)
      __builtin_prefetch(&A[(size_t)(m0 + ar) * K + k0 + 32 + ac], 0, 1);
      int pr = k0 + 32 + brw; if (pr > K - 1) pr = K - 1;
      __builtin_prefetch(&B[(size_t)pr * N + n0 + bcl], 0, 1);
    }
    __syncthreads();
    v8h alo = *(const v8h*)&sA[arow][half * 8];
    v8h ahi = *(const v8h*)&sA[arow][16 + half * 8];
    v16h af = CAT16(alo, ahi);
    v8h blo = *(const v8h*)&sBt[bcol][half * 8];
    v8h bhi = *(const v8h*)&sBt[bcol][16 + half * 8];
    v16h bf = CAT16(blo, bhi);
    acc = WMMA(af, bf, acc);
  }
#pragma unroll
  for (int r = 0; r < 8; ++r) {
    int m = m0 + wr * 16 + r + (half ? 8 : 0);
    int n = n0 + wc * 16 + lm;
    size_t o = (size_t)m * N + n;
    float v = acc[r];
    if (mode == 0) C[o] = v;
    else if (mode == 1) C[o] += v;
    else C[o] -= v;
  }
}

// epilogue: y = relu?( (C + bias) * g/sqrt(1+eps) + beta ) -> f16
__global__ void k_bias_act_h(const float* __restrict__ C, const float* __restrict__ bias,
                             const float* __restrict__ g, const float* __restrict__ beta,
                             f16* __restrict__ out, long M, int N, int relu) {
  long i = (long)blockIdx.x * blockDim.x + threadIdx.x;
  if (i >= M * N) return;
  int n = (int)(i % N);
  float v = C[i] + bias[n];
  if (g) v = v * (g[n] * rsqrtf(1.0f + 1e-5f)) + beta[n];
  if (relu) v = fmaxf(v, 0.0f);
  out[i] = (f16)v;
}

// feat = (zr+br)^2 + (zi+bi)^2
__global__ void k_feat(const float* __restrict__ Zr, const float* __restrict__ Zi,
                       const float* __restrict__ br, const float* __restrict__ bi,
                       float* __restrict__ feat, long M, int N) {
  long i = (long)blockIdx.x * blockDim.x + threadIdx.x;
  if (i >= M * N) return;
  int n = (int)(i % N);
  float a = Zr[i] + br[n], b = Zi[i] + bi[n];
  feat[i] = a * a + b * b;
}

// ---------------- FPS / ball query / gathers ----------------

__launch_bounds__(256)
__global__ void k_fps(const float* __restrict__ pts, int* __restrict__ fps_idx,
                      float* __restrict__ new_xyz) {
  __shared__ float dist[NP];
  __shared__ float rmax[256];
  __shared__ int rarg[256];
  __shared__ int far_s;
  int b = blockIdx.x, tid = threadIdx.x;
  for (int j = tid; j < NP; j += 256) dist[j] = 1e10f;
  if (tid == 0) far_s = 0;
  __syncthreads();
  for (int t = 0; t < NPT; ++t) {
    int far = far_s;
    if (tid == 0) fps_idx[b * NPT + t] = far;
    if (tid < 3) new_xyz[(b * NPT + t) * 3 + tid] = pts[((size_t)b * NP + far) * 3 + tid];
    float cx = pts[((size_t)b * NP + far) * 3 + 0];
    float cy = pts[((size_t)b * NP + far) * 3 + 1];
    float cz = pts[((size_t)b * NP + far) * 3 + 2];
    float lmx = -1.0f; int la = 0;
    for (int j = tid; j < NP; j += 256) {
      float dx = pts[((size_t)b * NP + j) * 3 + 0] - cx;
      float dy = pts[((size_t)b * NP + j) * 3 + 1] - cy;
      float dz = pts[((size_t)b * NP + j) * 3 + 2] - cz;
      float d = dx * dx + dy * dy + dz * dz;
      float nd = fminf(dist[j], d);
      dist[j] = nd;
      if (nd > lmx) { lmx = nd; la = j; }
    }
    rmax[tid] = lmx; rarg[tid] = la;
    __syncthreads();
    for (int s = 128; s > 0; s >>= 1) {
      if (tid < s) {
        if (rmax[tid + s] > rmax[tid] ||
            (rmax[tid + s] == rmax[tid] && rarg[tid + s] < rarg[tid])) {
          rmax[tid] = rmax[tid + s];
          rarg[tid] = rarg[tid + s];
        }
      }
      __syncthreads();
    }
    if (tid == 0) far_s = rarg[0];
    __syncthreads();
  }
}

__global__ void k_ball(const float* __restrict__ pts, const float* __restrict__ new_xyz,
                       int* __restrict__ ball) {
  int i = blockIdx.x * blockDim.x + threadIdx.x;  // NB*NPT
  if (i >= NB * NPT) return;
  int b = i / NPT;
  float cx = new_xyz[i * 3 + 0], cy = new_xyz[i * 3 + 1], cz = new_xyz[i * 3 + 2];
  int* out = ball + (size_t)i * NSAMP;
  int cnt = 0, first = -1;
  for (int j = 0; j < NP && cnt < NSAMP; ++j) {
    float dx = pts[((size_t)b * NP + j) * 3 + 0] - cx;
    float dy = pts[((size_t)b * NP + j) * 3 + 1] - cy;
    float dz = pts[((size_t)b * NP + j) * 3 + 2] - cz;
    float d = dx * dx + dy * dy + dz * dz;
    if (d <= 0.16f) {               // RADIUS^2
      if (first < 0) first = j;
      out[cnt++] = j;
    }
  }
  if (first < 0) first = 0;
  for (; cnt < NSAMP; ++cnt) out[cnt] = first;
}

// X0(b,s,k, 160) = [gxyz(3) | feat(128) | pad(29 zeros)] in f16
__global__ void k_gather(const float* __restrict__ pts, const float* __restrict__ new_xyz,
                         const float* __restrict__ feat, const int* __restrict__ ball,
                         f16* __restrict__ X0) {
  long i = (long)blockIdx.x * blockDim.x + threadIdx.x;  // NB*NPT*NSAMP*160
  if (i >= (long)NB * NPT * NSAMP * 160) return;
  int c = (int)(i % 160);
  long bsk = i / 160;
  int b = (int)(bsk / (NPT * NSAMP));
  int s = (int)((bsk / NSAMP) % NPT);
  int j = ball[bsk];
  float v;
  if (c < 3)       v = pts[((size_t)b * NP + j) * 3 + c] - new_xyz[(b * NPT + s) * 3 + c];
  else if (c < 131) v = feat[((size_t)b * NP + j) * 128 + (c - 3)];
  else              v = 0.0f;
  X0[i] = (f16)v;
}

// max over sample axis: in f16 (R,S,C) -> out f32 (R,C)
__global__ void k_maxpool(const f16* __restrict__ in, float* __restrict__ out,
                          int R, int S, int C) {
  long i = (long)blockIdx.x * blockDim.x + threadIdx.x;
  if (i >= (long)R * C) return;
  long r = i / C; int c = (int)(i % C);
  float m = -1e30f;
  for (int s = 0; s < S; ++s) m = fmaxf(m, (float)in[((size_t)r * S + s) * C + c]);
  out[i] = m;
}

// Xg(b*128+s, 288) = [new_xyz(3) | x2(256) | pad(29)]
__global__ void k_build_xg(const float* __restrict__ new_xyz, const float* __restrict__ x2,
                           f16* __restrict__ Xg) {
  int i = blockIdx.x * blockDim.x + threadIdx.x;  // 1024*288
  if (i >= NB * NPT * 288) return;
  int c = i % 288, r = i / 288;
  float v;
  if (c < 3)       v = new_xyz[r * 3 + c];
  else if (c < 259) v = x2[(size_t)r * 256 + (c - 3)];
  else              v = 0.0f;
  Xg[i] = (f16)v;
}

// ---------------- head (tiny M=8, plain dot kernels) ----------------

__global__ void k_fc(const float* __restrict__ X, const float* __restrict__ W,
                     const float* __restrict__ bias, const float* __restrict__ g,
                     const float* __restrict__ beta, float* __restrict__ out,
                     int M, int N, int K, int relu) {
  int i = blockIdx.x * blockDim.x + threadIdx.x;
  if (i >= M * N) return;
  int m = i / N, n = i % N;
  const float* x = X + (size_t)m * K;
  const float* w = W + (size_t)n * K;
  float acc = 0.0f;
  for (int k = 0; k < K; ++k) acc += x[k] * w[k];
  acc += bias[n];
  if (g) acc = acc * (g[n] * rsqrtf(1.0f + 1e-5f)) + beta[n];
  if (relu) acc = fmaxf(acc, 0.0f);
  out[i] = acc;
}

__global__ void k_logsoftmax(const float* __restrict__ logits, float* __restrict__ out, int N) {
  __shared__ float red[64];
  int b = blockIdx.x, t = threadIdx.x;
  float v = (t < N) ? logits[b * N + t] : -1e30f;
  red[t] = v;
  __syncthreads();
  for (int s = 32; s > 0; s >>= 1) { if (t < s) red[t] = fmaxf(red[t], red[t + s]); __syncthreads(); }
  float mx = red[0];
  __syncthreads();
  red[t] = (t < N) ? __expf(v - mx) : 0.0f;
  __syncthreads();
  for (int s = 32; s > 0; s >>= 1) { if (t < s) red[t] += red[t + s]; __syncthreads(); }
  float lse = logf(red[0]) + mx;
  if (t < N) out[b * N + t] = v - lse;
}

// ---------------- host orchestration ----------------

extern "C" void kernel_launch(void* const* d_in, const int* in_sizes, int n_in,
                              void* d_out, int out_size, void* d_ws, size_t ws_size,
                              hipStream_t stream) {
  (void)in_sizes; (void)n_in; (void)out_size; (void)ws_size;
  char* ws = (char*)d_ws;
  size_t cur = 0;
  auto take = [&](size_t bytes) -> void* {
    void* p = ws + cur;
    cur += (bytes + 255u) & ~(size_t)255u;
    return p;
  };
  auto ew = [](long n) { return dim3((unsigned)((n + 255) / 256)); };

  // ---- inputs ----
  const float* xyz = (const float*)d_in[0];
  const float* A   = (const float*)d_in[1];
  const float* Wr1 = (const float*)d_in[2];
  const float* Wi1 = (const float*)d_in[3];
  const float* br1 = (const float*)d_in[4];
  const float* bi1 = (const float*)d_in[5];
  const float* Wr2 = (const float*)d_in[6];
  const float* Wi2 = (const float*)d_in[7];
  const float* br2 = (const float*)d_in[8];
  const float* bi2 = (const float*)d_in[9];
  const float* s2w[3]  = {(const float*)d_in[10], (const float*)d_in[14], (const float*)d_in[18]};
  const float* s2b[3]  = {(const float*)d_in[11], (const float*)d_in[15], (const float*)d_in[19]};
  const float* s2g[3]  = {(const float*)d_in[12], (const float*)d_in[16], (const float*)d_in[20]};
  const float* s2be[3] = {(const float*)d_in[13], (const float*)d_in[17], (const float*)d_in[21]};
  const float* s3w[3]  = {(const float*)d_in[22], (const float*)d_in[26], (const float*)d_in[30]};
  const float* s3b[3]  = {(const float*)d_in[23], (const float*)d_in[27], (const float*)d_in[31]};
  const float* s3g[3]  = {(const float*)d_in[24], (const float*)d_in[28], (const float*)d_in[32]};
  const float* s3be[3] = {(const float*)d_in[25], (const float*)d_in[29], (const float*)d_in[33]};
  const float* fc1w = (const float*)d_in[34]; const float* fc1b = (const float*)d_in[35];
  const float* hg1  = (const float*)d_in[36]; const float* hb1  = (const float*)d_in[37];
  const float* fc2w = (const float*)d_in[38]; const float* fc2b = (const float*)d_in[39];
  const float* hg2  = (const float*)d_in[40]; const float* hb2  = (const float*)d_in[41];
  const float* fc3w = (const float*)d_in[42]; const float* fc3b = (const float*)d_in[43];

  // ---- workspace layout (with lifetime-based aliasing) ----
  const size_t BND = (size_t)NB * NP * DV;    // 4,194,304
  float* pts  = (float*)take((size_t)NB * NP * 3 * 4);
  float* eArF = (float*)take(BND * 4);
  float* eAiF = (float*)take(BND * 4);
  float* Mr   = (float*)take(BND * 4);
  float* Mi   = (float*)take(BND * 4);
  float* Cbuf = eArF;                         // 67 MB region reused for GEMM f32 outputs
  f16* eArH = (f16*)take(BND * 2);
  f16* eAiH = (f16*)take(BND * 2);
  char* act = (char*)take(50331648);          // activation alias region
  f16*   GrH  = (f16*)act;                          // 8,388,608 B
  f16*   GiH  = (f16*)(act + 8388608);              // 8,388,608 B
  f16*   arH  = (f16*)(act + 16777216);             // 4,194,304 B
  f16*   aiH  = (f16*)(act + 20971520);             // 4,194,304 B
  float* zr   = (float*)(act + 25165824);           // 8,388,608 B
  float* zi   = (float*)(act + 33554432);           // 8,388,608 B
  float* feat = (float*)(act + 41943040);           // 8,388,608 B
  f16* X0  = (f16*)act;                 // 20,971,520 B (G/a dead)
  f16* Hh1 = (f16*)(act + 20971520);    // 16,777,216 B (ai/zr/zi dead)
  f16* Hh2 = (f16*)act;                 // 16,777,216 B (X0 dead)
  f16* Hh3 = (f16*)(act + 16777216);    // 33,554,432 B (Hh1/feat dead)
  int*   fps_idx = (int*)take(NB * NPT * 4);
  float* new_xyz = (float*)take(NB * NPT * 3 * 4);
  int*   ball    = (int*)take((size_t)NB * NPT * NSAMP * 4);
  float* x2      = (float*)take((size_t)NB * NPT * 256 * 4);
  f16*   Xg      = (f16*)take((size_t)NB * NPT * 288 * 2);
  f16*   s3a     = (f16*)take((size_t)NB * NPT * 256 * 2);
  f16*   s3bh    = (f16*)take((size_t)NB * NPT * 512 * 2);
  f16*   s3c     = (f16*)take((size_t)NB * NPT * 1024 * 2);
  float* x3      = (float*)take(NB * 1024 * 4);
  float* h1      = (float*)take(NB * 512 * 4);
  float* h2      = (float*)take(NB * 256 * 4);
  float* logits  = (float*)take(NB * 40 * 4);
  f16* Wr1T = (f16*)take(256 * 128 * 2);
  f16* Wi1T = (f16*)take(256 * 128 * 2);
  f16* Wr2T = (f16*)take(128 * 128 * 2);
  f16* Wi2T = (f16*)take(128 * 128 * 2);
  f16* w20T = (f16*)take(160 * 128 * 2);
  f16* w21T = (f16*)take(128 * 128 * 2);
  f16* w22T = (f16*)take(128 * 256 * 2);
  f16* w30T = (f16*)take(288 * 256 * 2);
  f16* w31T = (f16*)take(256 * 512 * 2);
  f16* w32T = (f16*)take(512 * 1024 * 2);

  // ---- veckm ----
  k_transpose<<<ew(NB * NP), 256, 0, stream>>>(xyz, pts);
  k_eA<<<ew((long)NB * NP * DV), 256, 0, stream>>>(pts, A, eArF, eAiF, eArH, eAiH);
  k_veckm_gemm<<<dim3(NP / 16, 2, NB), 256, 0, stream>>>(pts, eArH, eAiH, Mr, Mi);
  k_gnorm<<<dim3(NB * NP), 256, 0, stream>>>(Mr, Mi, eArF, eAiF, GrH, GiH);

  // ---- complex classifier layers ----
  k_wT<<<ew(256 * 128), 256, 0, stream>>>(Wr1, Wr1T, 128, 256, 256);
  k_wT<<<ew(256 * 128), 256, 0, stream>>>(Wi1, Wi1T, 128, 256, 256);
  k_wT<<<ew(128 * 128), 256, 0, stream>>>(Wr2, Wr2T, 128, 128, 128);
  k_wT<<<ew(128 * 128), 256, 0, stream>>>(Wi2, Wi2T, 128, 128, 128);
  const int Mcl = NB * NP;  // 16384
  dim3 gcl(128 / 64, Mcl / 32);
  k_gemm<<<gcl, 256, 0, stream>>>(GrH, Wr1T, zr, Mcl, 128, 256, 0);
  k_gemm<<<gcl, 256, 0, stream>>>(GiH, Wi1T, zr, Mcl, 128, 256, 2);
  k_gemm<<<gcl, 256, 0, stream>>>(GrH, Wi1T, zi, Mcl, 128, 256, 0);
  k_gemm<<<gcl, 256, 0, stream>>>(GiH, Wr1T, zi, Mcl, 128, 256, 1);
  k_bias_act_h<<<ew((long)Mcl * 128), 256, 0, stream>>>(zr, br1, nullptr, nullptr, arH, Mcl, 128, 1);
  k_bias_act_h<<<ew((long)Mcl * 128), 256, 0, stream>>>(zi, bi1, nullptr, nullptr, aiH, Mcl, 128, 1);
  k_gemm<<<gcl, 256, 0, stream>>>(arH, Wr2T, zr, Mcl, 128, 128, 0);
  k_gemm<<<gcl, 256, 0, stream>>>(aiH, Wi2T, zr, Mcl, 128, 128, 2);
  k_gemm<<<gcl, 256, 0, stream>>>(arH, Wi2T, zi, Mcl, 128, 128, 0);
  k_gemm<<<gcl, 256, 0, stream>>>(aiH, Wr2T, zi, Mcl, 128, 128, 1);
  k_feat<<<ew((long)Mcl * 128), 256, 0, stream>>>(zr, zi, br2, bi2, feat, Mcl, 128);

  // ---- sampling / grouping ----
  k_fps<<<dim3(NB), 256, 0, stream>>>(pts, fps_idx, new_xyz);
  k_ball<<<ew(NB * NPT), 256, 0, stream>>>(pts, new_xyz, ball);
  k_gather<<<ew((long)NB * NPT * NSAMP * 160), 256, 0, stream>>>(pts, new_xyz, feat, ball, X0);

  // ---- sa2 MLP (M = 65536) ----
  k_wT<<<ew(160 * 128), 256, 0, stream>>>(s2w[0], w20T, 128, 131, 160);
  k_wT<<<ew(128 * 128), 256, 0, stream>>>(s2w[1], w21T, 128, 128, 128);
  k_wT<<<ew(128 * 256), 256, 0, stream>>>(s2w[2], w22T, 256, 128, 128);
  const int Ms2 = NB * NPT * NSAMP;  // 65536
  k_gemm<<<dim3(128 / 64, Ms2 / 32), 256, 0, stream>>>(X0, w20T, Cbuf, Ms2, 128, 160, 0);
  k_bias_act_h<<<ew((long)Ms2 * 128), 256, 0, stream>>>(Cbuf, s2b[0], s2g[0], s2be[0], Hh1, Ms2, 128, 1);
  k_gemm<<<dim3(128 / 64, Ms2 / 32), 256, 0, stream>>>(Hh1, w21T, Cbuf, Ms2, 128, 128, 0);
  k_bias_act_h<<<ew((long)Ms2 * 128), 256, 0, stream>>>(Cbuf, s2b[1], s2g[1], s2be[1], Hh2, Ms2, 128, 1);
  k_gemm<<<dim3(256 / 64, Ms2 / 32), 256, 0, stream>>>(Hh2, w22T, Cbuf, Ms2, 256, 128, 0);
  k_bias_act_h<<<ew((long)Ms2 * 256), 256, 0, stream>>>(Cbuf, s2b[2], s2g[2], s2be[2], Hh3, Ms2, 256, 1);
  k_maxpool<<<ew((long)NB * NPT * 256), 256, 0, stream>>>(Hh3, x2, NB * NPT, NSAMP, 256);

  // ---- sa3 MLP (M = 1024) ----
  k_build_xg<<<ew(NB * NPT * 288), 256, 0, stream>>>(new_xyz, x2, Xg);
  k_wT<<<ew(288 * 256), 256, 0, stream>>>(s3w[0], w30T, 256, 259, 288);
  k_wT<<<ew(256 * 512), 256, 0, stream>>>(s3w[1], w31T, 512, 256, 256);
  k_wT<<<ew(512 * 1024), 256, 0, stream>>>(s3w[2], w32T, 1024, 512, 512);
  const int Ms3 = NB * NPT;  // 1024
  k_gemm<<<dim3(256 / 64, Ms3 / 32), 256, 0, stream>>>(Xg, w30T, Cbuf, Ms3, 256, 288, 0);
  k_bias_act_h<<<ew((long)Ms3 * 256), 256, 0, stream>>>(Cbuf, s3b[0], s3g[0], s3be[0], s3a, Ms3, 256, 1);
  k_gemm<<<dim3(512 / 64, Ms3 / 32), 256, 0, stream>>>(s3a, w31T, Cbuf, Ms3, 512, 256, 0);
  k_bias_act_h<<<ew((long)Ms3 * 512), 256, 0, stream>>>(Cbuf, s3b[1], s3g[1], s3be[1], s3bh, Ms3, 512, 1);
  k_gemm<<<dim3(1024 / 64, Ms3 / 32), 256, 0, stream>>>(s3bh, w32T, Cbuf, Ms3, 1024, 512, 0);
  k_bias_act_h<<<ew((long)Ms3 * 1024), 256, 0, stream>>>(Cbuf, s3b[2], s3g[2], s3be[2], s3c, Ms3, 1024, 1);
  k_maxpool<<<ew((long)NB * 1024), 256, 0, stream>>>(s3c, x3, NB, NPT, 1024);

  // ---- head ----
  k_fc<<<ew(NB * 512), 256, 0, stream>>>(x3, fc1w, fc1b, hg1, hb1, h1, NB, 512, 1024, 1);
  k_fc<<<ew(NB * 256), 256, 0, stream>>>(h1, fc2w, fc2b, hg2, hb2, h2, NB, 256, 512, 1);
  k_fc<<<ew(NB * 40), 256, 0, stream>>>(h2, fc3w, fc3b, nullptr, nullptr, logits, NB, 40, 256, 0);
  k_logsoftmax<<<dim3(NB), 64, 0, stream>>>(logits, (float*)d_out, 40);
}
